// GAT_30288109371431
// MI455X (gfx1250) — matrix operand order repaired and checked by reading, not compiled, not run on hardware
//
#include <hip/hip_runtime.h>
#include <cstdint>
#include <cstddef>

// ---------------------------------------------------------------------------
// Model constants (match reference)
// ---------------------------------------------------------------------------
#define NEG_SLOPE 0.2f
#define BN_EPS    1e-5f

// ---------------------------------------------------------------------------
// WMMA types (gfx1250, wave32)
// ---------------------------------------------------------------------------
typedef __bf16 bf16_t;
typedef bf16_t v16bf __attribute__((ext_vector_type(16)));
typedef float  v8f   __attribute__((ext_vector_type(8)));

struct __attribute__((aligned(16))) U128 { unsigned int x, y, z, w; };

union ABFrag {
  v16bf v;
  U128 q[2];
  unsigned short s[16];
};

__device__ __forceinline__ unsigned short f32_to_bf16(float f) {
  unsigned u = __float_as_uint(f);
  u += 0x7FFFu + ((u >> 16) & 1u);          // round-to-nearest-even
  return (unsigned short)(u >> 16);
}

// order-preserving float<->uint map for atomicMax-based segment max
__device__ __forceinline__ unsigned encodeF(float f) {
  unsigned u = __float_as_uint(f);
  return (u & 0x80000000u) ? ~u : (u | 0x80000000u);
}
__device__ __forceinline__ float decodeF(unsigned e) {
  unsigned u = (e & 0x80000000u) ? (e & 0x7FFFFFFFu) : ~e;
  return __uint_as_float(u);
}

// ---------------------------------------------------------------------------
// Utility kernels
// ---------------------------------------------------------------------------
__global__ void zero_kernel(unsigned int* __restrict__ p, long n) {
  long t = (long)blockIdx.x * blockDim.x + threadIdx.x;
  if (t < n) p[t] = 0u;
}

// ---------------------------------------------------------------------------
// Operand packing into CDNA5 WMMA fragment order (ISA 7.12.2).
// A [M,K] f32 row-major -> bf16 fragments: one 32B fragment per (tileM,ks,lane)
//   lane<16 : row=tile*16+lane,   elems = A[row][k0+0..7]  ++ A[row][k0+16..23]
//   lane>=16: row=tile*16+lane-16 elems = A[row][k0+8..15] ++ A[row][k0+24..31]
// ---------------------------------------------------------------------------
__global__ void packA_kernel(const float* __restrict__ A,
                             unsigned short* __restrict__ Ap,
                             int M, int K) {
  const int KS = K / 32;
  const long total = (long)(M / 16) * KS * 32;
  long t = (long)blockIdx.x * blockDim.x + threadIdx.x;
  if (t >= total) return;
  const int lane = (int)(t & 31);
  const long fs = t >> 5;                 // tile*KS + ks
  const int tile = (int)(fs / KS), ks = (int)(fs % KS);
  const int row = tile * 16 + (lane & 15);
  const int base = ks * 32 + (lane >> 4) * 8;
  const float* ar = A + (size_t)row * K + base;
  ABFrag o;
#pragma unroll
  for (int e = 0; e < 8; ++e) o.s[e] = f32_to_bf16(ar[e]);
#pragma unroll
  for (int e = 0; e < 8; ++e) o.s[8 + e] = f32_to_bf16(ar[16 + e]);
  U128* dst = (U128*)Ap + t * 2;
  dst[0] = o.q[0];
  dst[1] = o.q[1];
}

// B [K,Nc] f32 row-major -> bf16 fragments: per (tileN,ks,lane):
//   n = tile*16 + (lane&15); k0 = ks*32 + (lane>>4)*8; elems = B[k0+e+(e>=8?8:0)][n]
__global__ void packB_kernel(const float* __restrict__ B,
                             unsigned short* __restrict__ Bp,
                             int K, int Nc) {
  const int KS = K / 32;
  const long total = (long)(Nc / 16) * KS * 32;
  long t = (long)blockIdx.x * blockDim.x + threadIdx.x;
  if (t >= total) return;
  const int lane = (int)(t & 31);
  const long fs = t >> 5;
  const int tile = (int)(fs / KS), ks = (int)(fs % KS);
  const int n = tile * 16 + (lane & 15);
  const int k0 = ks * 32 + (lane >> 4) * 8;
  ABFrag o;
#pragma unroll
  for (int e = 0; e < 16; ++e) o.s[e] = f32_to_bf16(B[(size_t)(k0 + e + (e >= 8 ? 8 : 0)) * Nc + n]);
  U128* dst = (U128*)Bp + t * 2;
  dst[0] = o.q[0];
  dst[1] = o.q[1];
}

// ---------------------------------------------------------------------------
// WMMA GEMM on pre-packed operands: C[M,Nc] = A[M,K] * B[K,Nc], C f32.
// One wave computes one 16x16 C tile; 4 waves/block cover 64 columns.
// K templated (=128 here) so the K-loop fully unrolls into 4 v_wmma ops.
// ---------------------------------------------------------------------------
template <int K>
__global__ void __launch_bounds__(128)
wmma_gemm_packed_kernel(const unsigned short* __restrict__ Ap,
                        const unsigned short* __restrict__ Bp,
                        float* __restrict__ C, int M, int Nc) {
  constexpr int KS = K / 32;
  const int lane  = threadIdx.x & 31;
  const int wave  = threadIdx.x >> 5;
  const int tileM = blockIdx.x;
  const int tileN = blockIdx.y * 4 + wave;

  const U128* A4 = (const U128*)Ap;
  const U128* B4 = (const U128*)Bp;
  const size_t abase = ((size_t)tileM * KS * 32 + lane) * 2;
  const size_t bbase = ((size_t)tileN * KS * 32 + lane) * 2;

  v8f acc = {};
#pragma unroll
  for (int ks = 0; ks < KS; ++ks) {
    ABFrag a, b;
    a.q[0] = A4[abase + (size_t)ks * 64];
    a.q[1] = A4[abase + (size_t)ks * 64 + 1];
    b.q[0] = B4[bbase + (size_t)ks * 64];
    b.q[1] = B4[bbase + (size_t)ks * 64 + 1];
    acc = __builtin_amdgcn_wmma_f32_16x16x32_bf16(
        /*neg_a=*/false, a.v, /*neg_b=*/false, b.v,
        /*c_mod=*/(short)0, acc, /*reuse_a=*/false, /*reuse_b=*/false);
  }

  const int n  = tileN * 16 + (lane & 15);
  const int hi = lane >> 4;
#pragma unroll
  for (int r = 0; r < 8; ++r) {
    const int row = tileM * 16 + r + hi * 8;
    C[(size_t)row * Nc + n] = acc[r];
  }
}

// ---------------------------------------------------------------------------
// Edge score: one wave per edge. score[e,h] = sum_c lrelu(xl[src]+xr[dst]+ee)*att
// ee computed on the fly: edge_attr[e,0:16] @ We[16,HC].
// ---------------------------------------------------------------------------
template <int HC, int NH>
__global__ void __launch_bounds__(256)
edge_score_kernel(const float* __restrict__ xl, const float* __restrict__ xr,
                  const float* __restrict__ eattr, const float* __restrict__ We,
                  const float* __restrict__ att, const int* __restrict__ src,
                  const int* __restrict__ dst, float* __restrict__ score, int E) {
  constexpr int CPL = HC / 32;      // channels per lane
  constexpr int CH  = HC / NH;      // channels per head
  constexpr int GRP = CH / CPL;     // lanes per head group
  const int wid  = (blockIdx.x * blockDim.x + threadIdx.x) >> 5;
  const int lane = threadIdx.x & 31;
  if (wid >= E) return;
  const int s = src[wid], d = dst[wid];
  const int c0 = lane * CPL;

  float ef[16];
#pragma unroll
  for (int f = 0; f < 16; ++f) ef[f] = eattr[(size_t)wid * 16 + f];

  const float* xls = xl + (size_t)s * HC;
  const float* xrd = xr + (size_t)d * HC;

  float accv = 0.f;
#pragma unroll
  for (int i = 0; i < CPL; ++i) {
    const int c = c0 + i;
    float ee = 0.f;
#pragma unroll
    for (int f = 0; f < 16; ++f) ee += ef[f] * We[f * HC + c];
    float mv = xls[c] + xrd[c] + ee;
    mv = (mv > 0.f) ? mv : NEG_SLOPE * mv;
    accv += mv * att[c];            // att flat [NH*CH] == global channel index
  }
#pragma unroll
  for (int off = GRP >> 1; off > 0; off >>= 1)
    accv += __shfl_xor(accv, off, 32);
  if ((lane % GRP) == 0) {
    const int h = c0 / CH;
    score[(size_t)wid * NH + h] = accv;
  }
}

__global__ void smax_kernel(const float* __restrict__ score,
                            const int* __restrict__ dst,
                            unsigned* __restrict__ smax_enc, long total, int NH) {
  long t = (long)blockIdx.x * blockDim.x + threadIdx.x;
  if (t >= total) return;
  const int e = (int)(t / NH), h = (int)(t % NH);
  atomicMax(&smax_enc[(size_t)dst[e] * NH + h], encodeF(score[t]));
}

__global__ void edge_exp_kernel(const float* __restrict__ score,
                                const int* __restrict__ dst,
                                const unsigned* __restrict__ smax_enc,
                                float* __restrict__ num, float* __restrict__ den,
                                long total, int NH) {
  long t = (long)blockIdx.x * blockDim.x + threadIdx.x;
  if (t >= total) return;
  const int e = (int)(t / NH), h = (int)(t % NH);
  const int d = dst[e];
  const float sm = decodeF(smax_enc[(size_t)d * NH + h]);
  const float v = __expf(score[t] - sm);
  num[t] = v;
  atomicAdd(&den[(size_t)d * NH + h], v);
}

template <int HC, int NH>
__global__ void __launch_bounds__(256)
edge_aggr_kernel(const float* __restrict__ xl, const float* __restrict__ num,
                 const float* __restrict__ den, const int* __restrict__ src,
                 const int* __restrict__ dst, float* __restrict__ out, int E) {
  constexpr int CPL = HC / 32;
  constexpr int CH  = HC / NH;
  const int wid  = (blockIdx.x * blockDim.x + threadIdx.x) >> 5;
  const int lane = threadIdx.x & 31;
  if (wid >= E) return;
  const int s = src[wid], d = dst[wid];
  const int c0 = lane * CPL;
  const float* xls = xl + (size_t)s * HC;
#pragma unroll
  for (int i = 0; i < CPL; ++i) {
    const int c = c0 + i;
    const int h = c / CH;
    const float alpha =
        num[(size_t)wid * NH + h] / (den[(size_t)d * NH + h] + 1e-16f);
    atomicAdd(&out[(size_t)d * HC + c], xls[c] * alpha);
  }
}

// bias + batchnorm(eval) + ELU [+ residual projection]
__global__ void node_post_kernel(const float* __restrict__ aggr,
                                 const float* __restrict__ bias,
                                 const float* __restrict__ gamma,
                                 const float* __restrict__ beta,
                                 const float* __restrict__ mean,
                                 const float* __restrict__ var,
                                 const float* __restrict__ resproj,
                                 const float* __restrict__ resb,
                                 float* __restrict__ hout, long total, int C) {
  long t = (long)blockIdx.x * blockDim.x + threadIdx.x;
  if (t >= total) return;
  const int c = (int)(t % C);
  float v = aggr[t] + bias[c];
  v = (v - mean[c]) * rsqrtf(var[c] + BN_EPS) * gamma[c] + beta[c];
  v = (v > 0.f) ? v : (__expf(v) - 1.f);
  if (resproj != nullptr) v += resproj[t] + resb[c];
  hout[t] = v;
}

__global__ void pool_kernel(const float* __restrict__ h,
                            const int* __restrict__ batch,
                            float* __restrict__ sums, float* __restrict__ cnt,
                            long total) {
  long t = (long)blockIdx.x * blockDim.x + threadIdx.x;
  if (t >= total) return;
  const int n = (int)(t / 64), c = (int)(t % 64);
  const int b = batch[n];
  atomicAdd(&sums[(size_t)b * 64 + c], h[t]);
  if (c == 0) atomicAdd(&cnt[b], 1.0f);
}

// predictor MLP: 64 -> 32 -> 16 -> 1, one thread per graph.
// Deliberately rolled loops: 256 threads total, static code size matters more
// than per-thread ILP here.
__global__ void __launch_bounds__(256)
mlp_kernel(const float* __restrict__ sums, const float* __restrict__ cnt,
           const float* __restrict__ W1, const float* __restrict__ b1,
           const float* __restrict__ W2, const float* __restrict__ b2,
           const float* __restrict__ W3, const float* __restrict__ b3,
           float* __restrict__ outp, int G) {
  const int g = blockIdx.x * blockDim.x + threadIdx.x;
  if (g >= G) return;
  float cn = cnt[g];
  cn = (cn > 1.0f) ? cn : 1.0f;
  const float inv = 1.0f / cn;
  const float* xs = sums + (size_t)g * 64;

  float h1[32];
  for (int j = 0; j < 32; ++j) {
    float a = b1[j];
    for (int c = 0; c < 64; ++c) a += xs[c] * inv * W1[c * 32 + j];
    h1[j] = (a > 0.f) ? a : 0.f;
  }
  float h2[16];
  for (int k = 0; k < 16; ++k) {
    float a = b2[k];
    for (int j = 0; j < 32; ++j) a += h1[j] * W2[j * 16 + k];
    h2[k] = (a > 0.f) ? a : 0.f;
  }
  float o = b3[0];
  for (int k = 0; k < 16; ++k) o += h2[k] * W3[k];
  outp[g] = o;
}

// ---------------------------------------------------------------------------
// Host-side orchestration
// ---------------------------------------------------------------------------
static inline long cdiv(long a, long b) { return (a + b - 1) / b; }

extern "C" void kernel_launch(void* const* d_in, const int* in_sizes, int n_in,
                              void* d_out, int out_size, void* d_ws, size_t ws_size,
                              hipStream_t stream) {
  (void)n_in; (void)ws_size;
  const int N = in_sizes[0] / 128;     // 50000
  const int E = in_sizes[1] / 2;       // 800000
  const int G = out_size;              // 256

  const float* x     = (const float*)d_in[0];
  const int*   eidx  = (const int*)d_in[1];
  const float* eattr = (const float*)d_in[2];
  const int*   batch = (const int*)d_in[3];
  const int* src = eidx;
  const int* dst = eidx + E;

  // params flattened in JAX pytree (sorted-key) order
  int p = 4;
  const float *bn_beta[3], *bn_gamma[3], *bn_mean[3], *bn_var[3];
  for (int i = 0; i < 3; ++i) {
    bn_beta[i]  = (const float*)d_in[p++];
    bn_gamma[i] = (const float*)d_in[p++];
    bn_mean[i]  = (const float*)d_in[p++];
    bn_var[i]   = (const float*)d_in[p++];
  }
  const float *gWe[3], *gWl[3], *gWr[3], *gAtt[3], *gBias[3];
  for (int i = 0; i < 3; ++i) {
    gWe[i]   = (const float*)d_in[p++];
    gWl[i]   = (const float*)d_in[p++];
    gWr[i]   = (const float*)d_in[p++];
    gAtt[i]  = (const float*)d_in[p++];
    gBias[i] = (const float*)d_in[p++];
  }
  const float* mW1 = (const float*)d_in[p++];
  const float* mW2 = (const float*)d_in[p++];
  const float* mW3 = (const float*)d_in[p++];
  const float* mb1 = (const float*)d_in[p++];
  const float* mb2 = (const float*)d_in[p++];
  const float* mb3 = (const float*)d_in[p++];
  const float* r1W = (const float*)d_in[p++];
  const float* r1b = (const float*)d_in[p++];
  const float* r2W = (const float*)d_in[p++];
  const float* r2b = (const float*)d_in[p++];

  // ---- workspace layout -------------------------------------------------
  char* ws = (char*)d_ws;
  size_t off = 0;
  auto take = [&](size_t bytes) -> char* {
    char* q = ws + off;
    off = (off + bytes + 255) & ~(size_t)255;
    return q;
  };
  float* xl      = (float*)take((size_t)N * 128 * 4);
  float* xr      = (float*)take((size_t)N * 128 * 4);
  float* h       = (float*)take((size_t)N * 128 * 4);
  float* aggr    = (float*)take((size_t)N * 128 * 4);
  float* resproj = (float*)take((size_t)N * 128 * 4);
  unsigned short* Abf = (unsigned short*)take((size_t)N * 128 * 2);  // packed A
  float* score   = (float*)take((size_t)E * 4 * 4);
  float* num     = (float*)take((size_t)E * 4 * 4);
  unsigned* smax = (unsigned*)take((size_t)N * 4 * 4);
  float* den     = (float*)take((size_t)N * 4 * 4);
  float* sums    = (float*)take((size_t)G * 64 * 4);
  float* cnt     = (float*)take((size_t)G * 4);
  // packed bf16 weights
  unsigned short* Wlb[3]; unsigned short* Wrb[3];
  unsigned short* R1bp; unsigned short* R2bp;
  const long wNc[3] = {128, 128, 64};
  for (int i = 0; i < 3; ++i) {
    Wlb[i] = (unsigned short*)take((size_t)128 * wNc[i] * 2);
    Wrb[i] = (unsigned short*)take((size_t)128 * wNc[i] * 2);
  }
  R1bp = (unsigned short*)take((size_t)128 * 128 * 2);
  R2bp = (unsigned short*)take((size_t)128 * 64 * 2);

  const int TB = 256;
  auto zero = [&](void* q, long n32) {
    zero_kernel<<<cdiv(n32, TB), TB, 0, stream>>>((unsigned int*)q, n32);
  };
  auto packA = [&](const float* s, unsigned short* d, int M, int K) {
    const long total = (long)(M / 16) * (K / 32) * 32;
    packA_kernel<<<cdiv(total, TB), TB, 0, stream>>>(s, d, M, K);
  };
  auto packB = [&](const float* s, unsigned short* d, int K, int Nc) {
    const long total = (long)(Nc / 16) * (K / 32) * 32;
    packB_kernel<<<cdiv(total, TB), TB, 0, stream>>>(s, d, K, Nc);
  };
  auto gemm = [&](const unsigned short* A, const unsigned short* B, float* C,
                  int M, int Nc) {   // K fixed at 128
    dim3 grid(M / 16, Nc / 64);
    wmma_gemm_packed_kernel<128><<<grid, 128, 0, stream>>>(A, B, C, M, Nc);
  };

  // pack all weights to fragment-ordered bf16
  for (int i = 0; i < 3; ++i) {
    packB(gWl[i], Wlb[i], 128, (int)wNc[i]);
    packB(gWr[i], Wrb[i], 128, (int)wNc[i]);
  }
  packB(r1W, R1bp, 128, 128);
  packB(r2W, R2bp, 128, 64);

  const long edgeWaveBlocks = cdiv((long)E * 32, TB);

  // ====================== layer 0 (128 -> 4x32) ==========================
  packA(x, Abf, N, 128);
  gemm(Abf, Wlb[0], xl, N, 128);
  gemm(Abf, Wrb[0], xr, N, 128);
  zero(smax, (long)N * 4); zero(den, (long)N * 4); zero(aggr, (long)N * 128);
  edge_score_kernel<128, 4><<<edgeWaveBlocks, TB, 0, stream>>>(
      xl, xr, eattr, gWe[0], gAtt[0], src, dst, score, E);
  smax_kernel<<<cdiv((long)E * 4, TB), TB, 0, stream>>>(score, dst, smax, (long)E * 4, 4);
  edge_exp_kernel<<<cdiv((long)E * 4, TB), TB, 0, stream>>>(score, dst, smax, num, den, (long)E * 4, 4);
  edge_aggr_kernel<128, 4><<<edgeWaveBlocks, TB, 0, stream>>>(xl, num, den, src, dst, aggr, E);
  node_post_kernel<<<cdiv((long)N * 128, TB), TB, 0, stream>>>(
      aggr, gBias[0], bn_gamma[0], bn_beta[0], bn_mean[0], bn_var[0],
      nullptr, nullptr, h, (long)N * 128, 128);

  // ====================== layer 1 (128 -> 4x32, residual) ================
  packA(h, Abf, N, 128);
  gemm(Abf, Wlb[1], xl, N, 128);
  gemm(Abf, Wrb[1], xr, N, 128);
  gemm(Abf, R1bp, resproj, N, 128);
  zero(smax, (long)N * 4); zero(den, (long)N * 4); zero(aggr, (long)N * 128);
  edge_score_kernel<128, 4><<<edgeWaveBlocks, TB, 0, stream>>>(
      xl, xr, eattr, gWe[1], gAtt[1], src, dst, score, E);
  smax_kernel<<<cdiv((long)E * 4, TB), TB, 0, stream>>>(score, dst, smax, (long)E * 4, 4);
  edge_exp_kernel<<<cdiv((long)E * 4, TB), TB, 0, stream>>>(score, dst, smax, num, den, (long)E * 4, 4);
  edge_aggr_kernel<128, 4><<<edgeWaveBlocks, TB, 0, stream>>>(xl, num, den, src, dst, aggr, E);
  node_post_kernel<<<cdiv((long)N * 128, TB), TB, 0, stream>>>(
      aggr, gBias[1], bn_gamma[1], bn_beta[1], bn_mean[1], bn_var[1],
      resproj, r1b, h, (long)N * 128, 128);

  // ====================== layer 2 (128 -> 1x64, residual) ================
  packA(h, Abf, N, 128);
  gemm(Abf, Wlb[2], xl, N, 64);
  gemm(Abf, Wrb[2], xr, N, 64);
  gemm(Abf, R2bp, resproj, N, 64);
  zero(smax, (long)N * 1); zero(den, (long)N * 1); zero(aggr, (long)N * 64);
  edge_score_kernel<64, 1><<<edgeWaveBlocks, TB, 0, stream>>>(
      xl, xr, eattr, gWe[2], gAtt[2], src, dst, score, E);
  smax_kernel<<<cdiv((long)E, TB), TB, 0, stream>>>(score, dst, smax, (long)E, 1);
  edge_exp_kernel<<<cdiv((long)E, TB), TB, 0, stream>>>(score, dst, smax, num, den, (long)E, 1);
  edge_aggr_kernel<64, 1><<<edgeWaveBlocks, TB, 0, stream>>>(xl, num, den, src, dst, aggr, E);
  node_post_kernel<<<cdiv((long)N * 64, TB), TB, 0, stream>>>(
      aggr, gBias[2], bn_gamma[2], bn_beta[2], bn_mean[2], bn_var[2],
      resproj, r2b, h, (long)N * 64, 64);

  // ====================== pool + MLP =====================================
  zero(sums, (long)G * 64); zero(cnt, (long)G);
  pool_kernel<<<cdiv((long)N * 64, TB), TB, 0, stream>>>(h, batch, sums, cnt, (long)N * 64);
  mlp_kernel<<<cdiv(G, 256), 256, 0, stream>>>(sums, cnt, mW1, mb1, mW2, mb2,
                                               mW3, mb3, (float*)d_out, G);
}
